// ModMultiHeadAttentionLayer_35785667510402
// MI455X (gfx1250) — compile-verified, matrix operand order checked by reading, not run on hardware
//
#include <hip/hip_runtime.h>
#include <hip/hip_bf16.h>

#define DEVINL __device__ __forceinline__

typedef __attribute__((ext_vector_type(16))) __bf16 bf16x16;
typedef __attribute__((ext_vector_type(8)))  __bf16 bf16x8;
typedef __attribute__((ext_vector_type(8)))  float  f32x8;

static constexpr int kB  = 8;
static constexpr int kS  = 1024;
static constexpr int kD  = 1024;
static constexpr int kH  = 16;
static constexpr int kHD = 64;
static constexpr int kCD = 512;
static constexpr int kBS = kB * kS;   // 8192 rows

// ---------------------------------------------------------------------------
// WMMA helpers
// ---------------------------------------------------------------------------
union FragU { bf16x16 v; bf16x8 h[2]; };

DEVINL f32x8 zero8() {
  f32x8 z = {0.f, 0.f, 0.f, 0.f, 0.f, 0.f, 0.f, 0.f};
  return z;
}

// Load a 16(row)x32(K) bf16 operand fragment from an LDS tile (row-major,
// leading dim `ld` elements, ld multiple of 8 so chunks stay 16B aligned).
// CDNA5 16-bit operand layout: lanes 0-15 row=lane, K chunks {0..7,16..23};
// lanes 16-31 same rows, K chunks {8..15,24..31}.
DEVINL bf16x16 lds_frag(const __bf16* tile, int ld, int row0, int lane) {
  const __bf16* p = tile + (row0 + (lane & 15)) * ld + ((lane >> 4) << 3);
  FragU f;
  f.h[0] = *(const bf16x8*)(p);
  f.h[1] = *(const bf16x8*)(p + 16);
  return f.v;
}

DEVINL f32x8 wmma_bf16f32(bf16x16 a, bf16x16 b, f32x8 c) {
  // (neg_a, A, neg_b, B, c_mod, C, reuse_a, reuse_b)
  return __builtin_amdgcn_wmma_f32_16x16x32_bf16(false, a, false, b,
                                                 (short)0, c, false, false);
}

// Async DMA of 16 bytes global -> LDS (per lane), tracked by ASYNCcnt.
// lds_off: wave-relative LDS byte address (low 32 bits of a generic pointer
// to a __shared__ object). gptr: 64-bit global address.
DEVINL void async_g2l_b128(unsigned lds_off, const void* gptr) {
  asm volatile("global_load_async_to_lds_b128 %0, %1, off"
               :: "v"(lds_off), "v"(gptr) : "memory");
}
DEVINL void wait_asynccnt0() {
  asm volatile("s_wait_asynccnt 0" ::: "memory");
}

// ---------------------------------------------------------------------------
// Block reductions (blockDim.x == 256, red[] has 256 floats)
// ---------------------------------------------------------------------------
DEVINL float block_reduce_sum(float v, float* red, int t) {
  red[t] = v; __syncthreads();
  for (int o = 128; o > 0; o >>= 1) {
    if (t < o) red[t] += red[t + o];
    __syncthreads();
  }
  float r = red[0];
  __syncthreads();
  return r;
}

DEVINL float block_reduce_max(float v, float* red, int t) {
  red[t] = v; __syncthreads();
  for (int o = 128; o > 0; o >>= 1) {
    if (t < o) red[t] = fmaxf(red[t], red[t + o]);
    __syncthreads();
  }
  float r = red[0];
  __syncthreads();
  return r;
}

// ---------------------------------------------------------------------------
// Kernel 1: tc_p = LayerNorm(code @ wc_w.T + wc_b)  for p in {q,k,v,o}
// ---------------------------------------------------------------------------
struct TcArgs {
  const float* code;
  const float* wc_w[4];
  const float* wc_b[4];
  const float* ln_g[4];
  const float* ln_b[4];
  float* tc_out;   // [4][kD]
};

__global__ void __launch_bounds__(256) tc_kernel(TcArgs A) {
  __shared__ float s_code[kCD];
  __shared__ float s_red[256];
  const int p = blockIdx.x;
  const int t = threadIdx.x;
  for (int i = t; i < kCD; i += 256) s_code[i] = A.code[i];
  __syncthreads();

  const float* wcw = A.wc_w[p];
  const float* wcb = A.wc_b[p];
  float y[4];
  float s1 = 0.f, s2 = 0.f;
  #pragma unroll
  for (int r = 0; r < 4; ++r) {
    const int j = t * 4 + r;
    const float* wr = wcw + j * kCD;
    float acc = 0.f;
    for (int k = 0; k < kCD; k += 4) {
      acc += wr[k + 0] * s_code[k + 0];
      acc += wr[k + 1] * s_code[k + 1];
      acc += wr[k + 2] * s_code[k + 2];
      acc += wr[k + 3] * s_code[k + 3];
    }
    acc += wcb[j];
    y[r] = acc;
    s1 += acc;
    s2 += acc * acc;
  }
  const float sum   = block_reduce_sum(s1, s_red, t);
  const float sumsq = block_reduce_sum(s2, s_red, t);
  const float mean  = sum * (1.f / kD);
  const float var   = sumsq * (1.f / kD) - mean * mean;
  const float rstd  = rsqrtf(var + 1e-5f);
  float* out = A.tc_out + p * kD;
  #pragma unroll
  for (int r = 0; r < 4; ++r) {
    const int j = t * 4 + r;
    out[j] = (y[r] - mean) * rstd * A.ln_g[p][j] + A.ln_b[p][j];
  }
}

// ---------------------------------------------------------------------------
// Kernel 2/6: modulated GEMM  Y[i,j] = sum_d (tc[d]*x[i,d]) * W[j,d] + b[j]
// Block tile 128(M)x128(N), K-chunks of 32; 8 waves arranged 2(M)x4(N),
// each wave owns 64x32 = 4x2 WMMA accumulators.
// Software-pipelined: global loads for K-chunk k+1 are issued before the
// WMMA block of chunk k, so the loads ride out behind 8 WMMAs.
// ---------------------------------------------------------------------------
struct GemmArgs {
  const float* x[3];     // [kBS, kD] inputs, selected by blockIdx.z
  const float* tc[3];    // [kD] modulation vectors
  const float* W;        // [kD, kD]
  const float* bias;     // [kD]
  __bf16* outb[3];       // bf16 outputs (Q/K/V path) or nullptr
  float*  outf;          // f32 output (final O path) or nullptr
};

__global__ void __launch_bounds__(256) modlin_gemm(GemmArgs A) {
  constexpr int LD = 40;                  // 32 + 8 pad (keeps 16B alignment)
  __shared__ __bf16 sA[128 * LD];
  __shared__ __bf16 sB[128 * LD];

  const int z  = blockIdx.z;
  const float* x  = A.x[z];
  const float* tc = A.tc[z];
  const int n0 = blockIdx.x * 128;
  const int m0 = blockIdx.y * 128;
  const int t = threadIdx.x, lane = t & 31, wid = t >> 5;
  const int wm = wid & 1, wn = wid >> 1;

  int rrow[4], rcol[4];
  #pragma unroll
  for (int it = 0; it < 4; ++it) {
    const int idx = t + it * 256;         // 0..1023
    rrow[it] = idx >> 3;                  // 0..127
    rcol[it] = (idx & 7) * 4;             // 0..28
  }

  float4 rx[4], rw[4], rt[4];
  auto load_stage = [&](int k0) {
    #pragma unroll
    for (int it = 0; it < 4; ++it) {
      rt[it] = *(const float4*)(tc + k0 + rcol[it]);
      rx[it] = *(const float4*)(x + (size_t)(m0 + rrow[it]) * kD + k0 + rcol[it]);
      rw[it] = *(const float4*)(A.W + (size_t)(n0 + rrow[it]) * kD + k0 + rcol[it]);
    }
  };

  f32x8 acc[4][2];
  #pragma unroll
  for (int i = 0; i < 4; ++i)
    #pragma unroll
    for (int n = 0; n < 2; ++n) acc[i][n] = zero8();

  load_stage(0);
  for (int k0 = 0; k0 < kD; k0 += 32) {
    __syncthreads();                      // prior compute done reading LDS
    #pragma unroll
    for (int it = 0; it < 4; ++it) {
      __bf16* da = sA + rrow[it] * LD + rcol[it];
      da[0] = (__bf16)(rx[it].x * rt[it].x);
      da[1] = (__bf16)(rx[it].y * rt[it].y);
      da[2] = (__bf16)(rx[it].z * rt[it].z);
      da[3] = (__bf16)(rx[it].w * rt[it].w);
      __bf16* db = sB + rrow[it] * LD + rcol[it];
      db[0] = (__bf16)rw[it].x; db[1] = (__bf16)rw[it].y;
      db[2] = (__bf16)rw[it].z; db[3] = (__bf16)rw[it].w;
    }
    __syncthreads();
    if (k0 + 32 < kD) load_stage(k0 + 32);  // issue next tile early

    bf16x16 fa[4], fb[2];
    #pragma unroll
    for (int i = 0; i < 4; ++i) fa[i] = lds_frag(sA, LD, wm * 64 + i * 16, lane);
    #pragma unroll
    for (int n = 0; n < 2; ++n) fb[n] = lds_frag(sB, LD, wn * 32 + n * 16, lane);
    #pragma unroll
    for (int i = 0; i < 4; ++i)
      #pragma unroll
      for (int n = 0; n < 2; ++n)
        acc[i][n] = wmma_bf16f32(fa[i], fb[n], acc[i][n]);
  }

  // Epilogue. C layout: lanes 0-15 col=lane, M=vgpr; lanes 16-31 col=lane-16,
  // M=vgpr+8.
  const int cn = lane & 15, mofs = (lane >> 4) * 8;
  if (A.outf != nullptr) {
    float* out = A.outf;
    #pragma unroll
    for (int i = 0; i < 4; ++i)
      #pragma unroll
      for (int n = 0; n < 2; ++n) {
        const int col = n0 + wn * 32 + n * 16 + cn;
        const float bv = A.bias[col];
        #pragma unroll
        for (int r = 0; r < 8; ++r) {
          const int row = m0 + wm * 64 + i * 16 + mofs + r;
          out[(size_t)row * kD + col] = acc[i][n][r] + bv;
        }
      }
  } else {
    __bf16* out = A.outb[z];
    #pragma unroll
    for (int i = 0; i < 4; ++i)
      #pragma unroll
      for (int n = 0; n < 2; ++n) {
        const int col = n0 + wn * 32 + n * 16 + cn;
        const float bv = A.bias[col];
        #pragma unroll
        for (int r = 0; r < 8; ++r) {
          const int row = m0 + wm * 64 + i * 16 + mofs + r;
          out[(size_t)row * kD + col] = (__bf16)(acc[i][n][r] + bv);
        }
      }
  }
}

// ---------------------------------------------------------------------------
// Kernel 3: raw energy  E[bh,i,j] = (1/8) * sum_d Q[b,i,h*64+d] K[b,j,h*64+d]
// Block tile 128x128, K=64 staged fully in LDS via async global->LDS DMA
// (bf16 operands, no transformation needed -> no VGPR round-trip).
// ---------------------------------------------------------------------------
__global__ void __launch_bounds__(256)
energy_kernel(const __bf16* __restrict__ Qb, const __bf16* __restrict__ Kb,
              float* __restrict__ attn) {
  constexpr int LD = 72;                  // 64 + 8 pad
  __shared__ __bf16 sQ[128 * LD];
  __shared__ __bf16 sK[128 * LD];
  const int bh = blockIdx.z;
  const int b  = bh >> 4;                 // kH == 16
  const int h  = bh & 15;
  const int j0 = blockIdx.x * 128;
  const int i0 = blockIdx.y * 128;
  const int t = threadIdx.x, lane = t & 31, wid = t >> 5;
  const size_t base = (size_t)b * kS * kD + (size_t)h * kHD;

  #pragma unroll
  for (int it = 0; it < 4; ++it) {
    const int idx = t + it * 256;         // 0..1023
    const int r   = idx >> 3;             // 0..127
    const int c8  = (idx & 7) * 8;        // 0..56
    const unsigned lq = (unsigned)(size_t)(sQ + r * LD + c8);
    const unsigned lk = (unsigned)(size_t)(sK + r * LD + c8);
    async_g2l_b128(lq, Qb + base + (size_t)(i0 + r) * kD + c8);
    async_g2l_b128(lk, Kb + base + (size_t)(j0 + r) * kD + c8);
  }
  wait_asynccnt0();                        // our DMAs landed in LDS
  __syncthreads();                         // everyone's DMAs visible

  const int wm = wid & 1, wn = wid >> 1;
  f32x8 acc[4][2];
  #pragma unroll
  for (int i = 0; i < 4; ++i)
    #pragma unroll
    for (int n = 0; n < 2; ++n) acc[i][n] = zero8();

  #pragma unroll
  for (int kk = 0; kk < kHD; kk += 32) {
    bf16x16 fa[4], fb[2];
    #pragma unroll
    for (int i = 0; i < 4; ++i) fa[i] = lds_frag(sQ + kk, LD, wm * 64 + i * 16, lane);
    #pragma unroll
    for (int n = 0; n < 2; ++n) fb[n] = lds_frag(sK + kk, LD, wn * 32 + n * 16, lane);
    #pragma unroll
    for (int i = 0; i < 4; ++i)
      #pragma unroll
      for (int n = 0; n < 2; ++n)
        acc[i][n] = wmma_bf16f32(fa[i], fb[n], acc[i][n]);
  }

  float* out = attn + (size_t)bh * kS * kS;
  const int cn = lane & 15, mofs = (lane >> 4) * 8;
  #pragma unroll
  for (int i = 0; i < 4; ++i)
    #pragma unroll
    for (int n = 0; n < 2; ++n) {
      const int col = j0 + wn * 32 + n * 16 + cn;
      #pragma unroll
      for (int r = 0; r < 8; ++r) {
        const int row = i0 + wm * 64 + i * 16 + mofs + r;
        out[(size_t)row * kS + col] = acc[i][n][r] * 0.125f;  // 1/sqrt(64)
      }
    }
}

// ---------------------------------------------------------------------------
// Kernel 4: per-row softmax stats (max, 1/sum(exp)). One block per row.
// ---------------------------------------------------------------------------
__global__ void __launch_bounds__(256)
rowstat_kernel(const float* __restrict__ attn, float2* __restrict__ stats) {
  __shared__ float s_red[256];
  const size_t row = blockIdx.x;          // 0 .. B*H*S-1
  const float* e = attn + row * kS;
  const int t = threadIdx.x;
  float v[4];
  float mx = -3.402823e38f;
  #pragma unroll
  for (int r = 0; r < 4; ++r) { v[r] = e[t + 256 * r]; mx = fmaxf(mx, v[r]); }
  mx = block_reduce_max(mx, s_red, t);
  float s = 0.f;
  #pragma unroll
  for (int r = 0; r < 4; ++r) s += __expf(v[r] - mx);
  s = block_reduce_sum(s, s_red, t);
  if (t == 0) {
    float2 o; o.x = mx; o.y = 1.f / s;
    stats[row] = o;
  }
}

// ---------------------------------------------------------------------------
// Kernel 5 (fused): softmax + u_i*u_j scale + binarize a/(1e-6+a),
// write final attention to d_out, and X = P @ V via WMMA.
// Block: 128 rows of one (b,h); loops over all j in chunks of 32.
// 8 waves, each owns 16(i) x 64(d) = 4 accumulators.
// Software-pipelined: raw-energy reads for chunk j+1 are issued before the
// WMMA block of chunk j.
// ---------------------------------------------------------------------------
__global__ void __launch_bounds__(256)
pv_kernel(float* __restrict__ attn, const float2* __restrict__ stats,
          const float* __restrict__ u, const __bf16* __restrict__ Vb,
          float* __restrict__ Xo) {
  constexpr int LDP = 40;
  constexpr int LDV = 40;
  __shared__ __bf16 sP[128 * LDP];
  __shared__ __bf16 sV[kHD * LDV];        // transposed: [d][j]
  const int bh = blockIdx.y;
  const int b  = bh >> 4;
  const int h  = bh & 15;
  const int i0 = blockIdx.x * 128;
  const int t = threadIdx.x, lane = t & 31, wid = t >> 5;
  float* P = attn + (size_t)bh * kS * kS;
  const float2* st = stats + (size_t)bh * kS;
  const float* ub = u + b * kS;
  const size_t vbase = (size_t)b * kS * kD + (size_t)h * kHD;

  int rrow[4], rcol[4];
  #pragma unroll
  for (int it = 0; it < 4; ++it) {
    const int idx = t + it * 256;
    rrow[it] = idx >> 3;                  // 0..127
    rcol[it] = (idx & 7) * 4;             // 0..28
  }

  float4 pe[4];
  auto load_p = [&](int j0) {
    #pragma unroll
    for (int it = 0; it < 4; ++it)
      pe[it] = *(const float4*)(P + (size_t)(i0 + rrow[it]) * kS + j0 + rcol[it]);
  };

  f32x8 acc[4];
  #pragma unroll
  for (int n = 0; n < 4; ++n) acc[n] = zero8();

  load_p(0);
  for (int j0 = 0; j0 < kS; j0 += 32) {
    __syncthreads();
    // --- softmax/binarize 128x32 tile of P; write final attention; stage bf16
    #pragma unroll
    for (int it = 0; it < 4; ++it) {
      const int r   = rrow[it];
      const int c4  = rcol[it];
      const int row = i0 + r;
      const float2 s = st[row];
      const float ui = ub[row];
      float a0 = ui * ub[j0 + c4 + 0] * __expf(pe[it].x - s.x) * s.y;
      float a1 = ui * ub[j0 + c4 + 1] * __expf(pe[it].y - s.x) * s.y;
      float a2 = ui * ub[j0 + c4 + 2] * __expf(pe[it].z - s.x) * s.y;
      float a3 = ui * ub[j0 + c4 + 3] * __expf(pe[it].w - s.x) * s.y;
      float4 av;
      av.x = a0 / (1e-6f + a0);
      av.y = a1 / (1e-6f + a1);
      av.z = a2 / (1e-6f + a2);
      av.w = a3 / (1e-6f + a3);
      *(float4*)(P + (size_t)row * kS + j0 + c4) = av;   // final attention out
      __bf16* d = sP + r * LDP + c4;
      d[0] = (__bf16)av.x; d[1] = (__bf16)av.y;
      d[2] = (__bf16)av.z; d[3] = (__bf16)av.w;
    }
    // --- V tile (32 j-rows x 64 d) -> LDS transposed [d][j]
    {
      const int r  = t >> 3;              // 0..31
      const int c8 = (t & 7) * 8;         // 0..56
      bf16x8 vv = *(const bf16x8*)(Vb + vbase + (size_t)(j0 + r) * kD + c8);
      #pragma unroll
      for (int q = 0; q < 8; ++q) sV[(c8 + q) * LDV + r] = vv[q];
    }
    __syncthreads();
    if (j0 + 32 < kS) load_p(j0 + 32);    // issue next energy tile early

    const bf16x16 fa = lds_frag(sP, LDP, wid * 16, lane);
    #pragma unroll
    for (int n = 0; n < 4; ++n) {
      const bf16x16 fb = lds_frag(sV, LDV, n * 16, lane);
      acc[n] = wmma_bf16f32(fa, fb, acc[n]);
    }
  }

  const int cn = lane & 15, mofs = (lane >> 4) * 8;
  float* out = Xo + (size_t)b * kS * kD + (size_t)h * kHD;
  #pragma unroll
  for (int n = 0; n < 4; ++n)
    #pragma unroll
    for (int r = 0; r < 8; ++r) {
      const int row = i0 + wid * 16 + mofs + r;
      out[(size_t)row * kD + n * 16 + cn] = acc[n][r];
    }
}

// ---------------------------------------------------------------------------
// Launcher
// ---------------------------------------------------------------------------
extern "C" void kernel_launch(void* const* d_in, const int* in_sizes, int n_in,
                              void* d_out, int out_size, void* d_ws, size_t ws_size,
                              hipStream_t stream) {
  (void)in_sizes; (void)n_in; (void)out_size; (void)ws_size;

  const float* query = (const float*)d_in[0];
  const float* key   = (const float*)d_in[1];
  const float* value = (const float*)d_in[2];
  const float* u     = (const float*)d_in[3];
  const float* code  = (const float*)d_in[4];
  const float* W     = (const float*)d_in[5];
  const float* bias  = (const float*)d_in[6];

  // Workspace layout (bytes):
  //   [0,      16K)  tc_all  (4 x 1024 f32)
  //   [1M,      2M)  stats   (B*H*S float2 = 1 MB)
  //   [2M,     50M)  Qb,Kb,Vb bf16 (16 MB each)
  //   [50M,    82M)  Xattn f32 (32 MB)
  char* ws = (char*)d_ws;
  float*  tc_all = (float*)ws;
  float2* stats  = (float2*)(ws + (1ull << 20));
  __bf16* Qb = (__bf16*)(ws + (2ull << 20));
  __bf16* Kb = Qb + (size_t)kBS * kD;
  __bf16* Vb = Kb + (size_t)kBS * kD;
  float*  Xattn = (float*)(ws + (2ull << 20) + 3ull * kBS * kD * sizeof(__bf16));

  float* xout = (float*)d_out;                       // [B,S,D]
  float* attn = xout + (size_t)kBS * kD;             // [B,H,S,S]

  // --- 1: transformed codes
  TcArgs ta;
  ta.code = code;
  ta.tc_out = tc_all;
  for (int p = 0; p < 4; ++p) {
    ta.wc_w[p] = (const float*)d_in[7 + p * 4 + 0];
    ta.wc_b[p] = (const float*)d_in[7 + p * 4 + 1];
    ta.ln_g[p] = (const float*)d_in[7 + p * 4 + 2];
    ta.ln_b[p] = (const float*)d_in[7 + p * 4 + 3];
  }
  tc_kernel<<<dim3(4), dim3(256), 0, stream>>>(ta);

  // --- 2: Q/K/V modulated GEMMs -> bf16
  GemmArgs ga;
  ga.x[0] = query; ga.x[1] = key; ga.x[2] = value;
  ga.tc[0] = tc_all; ga.tc[1] = tc_all + kD; ga.tc[2] = tc_all + 2 * kD;
  ga.W = W; ga.bias = bias;
  ga.outb[0] = Qb; ga.outb[1] = Kb; ga.outb[2] = Vb;
  ga.outf = nullptr;
  modlin_gemm<<<dim3(kD / 128, kBS / 128, 3), dim3(256), 0, stream>>>(ga);

  // --- 3: raw energy -> attention region of d_out (scratch for now)
  energy_kernel<<<dim3(kS / 128, kS / 128, kB * kH), dim3(256), 0, stream>>>(
      Qb, Kb, attn);

  // --- 4: per-row softmax stats
  rowstat_kernel<<<dim3(kB * kH * kS), dim3(256), 0, stream>>>(attn, stats);

  // --- 5: fused softmax/binarize + attention write + P@V
  pv_kernel<<<dim3(kS / 128, kB * kH), dim3(256), 0, stream>>>(
      attn, stats, u, Vb, Xattn);

  // --- 6: final modulated GEMM -> x output (f32)
  GemmArgs go;
  go.x[0] = Xattn; go.x[1] = Xattn; go.x[2] = Xattn;
  go.tc[0] = tc_all + 3 * kD; go.tc[1] = go.tc[0]; go.tc[2] = go.tc[0];
  go.W = W; go.bias = bias;
  go.outb[0] = nullptr; go.outb[1] = nullptr; go.outb[2] = nullptr;
  go.outf = xout;
  modlin_gemm<<<dim3(kD / 128, kBS / 128, 1), dim3(256), 0, stream>>>(go);
}
